// CurvedMultiHeadAttention_14972255994029
// MI455X (gfx1250) — matrix operand order repaired and checked by reading, not compiled, not run on hardware
//
// CurvedMultiHeadAttention for MI455X (gfx1250, wave32, WMMA).
// Compute-bound (~35 GFLOP vs ~35MB traffic @23.3TB/s) -> all GEMMs on
// v_wmma_f32_16x16x32_f16 (f16 operands, f32 accumulate). Flash-style
// attention (online softmax), no SxS materialization. Tile staging uses
// CDNA5 async DMA (global_load_async_to_lds_b128, ASYNCcnt) with LDS
// double buffering so the DMA of tile k+1 overlaps WMMA on tile k.
#include <hip/hip_runtime.h>
#include <hip/hip_bf16.h>

#define EMBED 768
#define HEADS 12
#define HDIM  64
#define RANK  16
#define BATCH 4
#define SEQ   1024
#define BH    (BATCH*HEADS)
#define MROWS (BATCH*SEQ)
#define EPSV  1e-6f
#define SCALE 0.125f   // 1/sqrt(64)

typedef __attribute__((ext_vector_type(16))) _Float16 v16h;
typedef __attribute__((ext_vector_type(8)))  float    v8f;

#define LDH 72   // LDS tile pitch in halfs: 144B rows (16B-aligned, staggers banks)

__device__ __forceinline__ v8f wmma_f16(v16h a, v16h b, v8f c) {
  return __builtin_amdgcn_wmma_f32_16x16x32_f16(false, a, false, b, (short)0, c,
                                                false, false);
}

// A fragment (16x32 f16) from LDS tile stored row-major [m][k], pitch LDH.
// ISA layout: lane m=lane&15; VGPR j<4 holds K=2j+8*(lane>>4)(+1),
// VGPR j>=4 holds K=16+2(j-4)+8*(lane>>4)(+1)  -> two contiguous 16B loads.
__device__ __forceinline__ v16h frag_a(const _Float16* tile, int koff) {
  const int lane = threadIdx.x & 31;
  const _Float16* row = tile + (lane & 15) * LDH + koff + ((lane >> 4) << 3);
  v16h a;
  ((uint4*)&a)[0] = *(const uint4*)(row);
  ((uint4*)&a)[1] = *(const uint4*)(row + 16);
  return a;
}

// B fragment (32x16 f16, B[k][n]) from LDS tile stored [n][k] row-major.
// ISA layout: lane n=lane&15; e<8 holds K=e+8*(lane>>4), e>=8 holds K=16+(e-8)+8*(lane>>4).
__device__ __forceinline__ v16h frag_b(const _Float16* tile, int koff) {
  const int lane = threadIdx.x & 31;
  const _Float16* row = tile + (lane & 15) * LDH + koff + ((lane >> 4) << 3);
  v16h b;
  ((uint4*)&b)[0] = *(const uint4*)(row);
  ((uint4*)&b)[1] = *(const uint4*)(row + 16);
  return b;
}

// Async-DMA a 64x64 f16 tile (global, row stride sstride halfs) into LDS
// [64][LDH] without touching VGPRs for the data: 512 16B segments over
// 128 threads -> 4 global_load_async_to_lds_b128 per thread (ASYNCcnt).
// LDS operand = wave-relative LDS byte offset = low 32 bits of the generic
// pointer to a __shared__ object (FLAT aperture rule, ISA 10.2).
__device__ __forceinline__ void async_stage_tile(_Float16* dst,
                                                 const _Float16* src,
                                                 int sstride) {
#pragma unroll
  for (int j = 0; j < 4; ++j) {
    int i = threadIdx.x + 128 * j;
    int r = i >> 3, seg = (i & 7) << 3;
    const _Float16* g = src + (size_t)r * sstride + seg;
    unsigned l = (unsigned)(uintptr_t)(dst + r * LDH + seg);
    asm volatile("global_load_async_to_lds_b128 %0, %1, off"
                 :: "v"(l), "v"(g)
                 : "memory");
  }
}

__device__ __forceinline__ void wait_async() {
  asm volatile("s_wait_asynccnt 0x0" ::: "memory");
}

// ---------------- elementwise f32 -> f16 ----------------
__global__ void cvt_f32_f16(const float* __restrict__ src,
                            _Float16* __restrict__ dst, int n) {
  int i = blockIdx.x * blockDim.x + threadIdx.x;
  if (i < n) dst[i] = (_Float16)src[i];
}

// ---------------- G_h = A_h A_h^T + eps*I (f32 math, f16 store) ----------------
__global__ void build_g(const float* __restrict__ A, _Float16* __restrict__ Gh) {
  int h = blockIdx.x;
  const float* Ah = A + (size_t)h * HDIM * RANK;
  for (int idx = threadIdx.x; idx < HDIM * HDIM; idx += blockDim.x) {
    int i = idx >> 6, j = idx & 63;
    float s = (i == j) ? EPSV : 0.f;
#pragma unroll
    for (int r = 0; r < RANK; ++r) s += Ah[i * RANK + r] * Ah[j * RANK + r];
    Gh[(size_t)h * HDIM * HDIM + idx] = (_Float16)s;
  }
}

// ---------------- QKV projection: Y = X @ W^T + b, f16 out in [b,h,s,d] ----------------
__global__ void __launch_bounds__(128)
gemm_qkv(const _Float16* __restrict__ Xh,
         const _Float16* __restrict__ Wq, const _Float16* __restrict__ Wk,
         const _Float16* __restrict__ Wv,
         const float* __restrict__ bq, const float* __restrict__ bk,
         const float* __restrict__ bv,
         _Float16* __restrict__ Q, _Float16* __restrict__ K,
         _Float16* __restrict__ V) {
  __shared__ __align__(16) _Float16 Xs[2][64 * LDH];
  __shared__ __align__(16) _Float16 Ws[2][64 * LDH];
  const int h  = blockIdx.x;            // 64 output cols == one head
  const int m0 = blockIdx.y * 64;
  const _Float16* W; const float* bias; _Float16* dst;
  if (blockIdx.z == 0)      { W = Wq; bias = bq; dst = Q; }
  else if (blockIdx.z == 1) { W = Wk; bias = bk; dst = K; }
  else                      { W = Wv; bias = bv; dst = V; }
  const int wid  = threadIdx.x >> 5;
  const int lane = threadIdx.x & 31;
  const int colb = lane & 15, rowb = (lane >> 4) * 8;

  async_stage_tile(Xs[0], Xh + (size_t)m0 * EMBED, EMBED);
  async_stage_tile(Ws[0], W + (size_t)h * 64 * EMBED, EMBED);
  wait_async();
  __syncthreads();

  v8f acc[4] = {};
#pragma unroll
  for (int kb = 0; kb < EMBED; kb += 64) {
    const int buf = (kb >> 6) & 1;
    if (kb + 64 < EMBED) {     // DMA next tile while WMMA consumes this one
      async_stage_tile(Xs[buf ^ 1], Xh + (size_t)m0 * EMBED + kb + 64, EMBED);
      async_stage_tile(Ws[buf ^ 1], W + (size_t)h * 64 * EMBED + kb + 64, EMBED);
    }
    v16h a0 = frag_a(Xs[buf] + wid * 16 * LDH, 0);
    v16h a1 = frag_a(Xs[buf] + wid * 16 * LDH, 32);
#pragma unroll
    for (int nt = 0; nt < 4; ++nt) {
      acc[nt] = wmma_f16(a0, frag_b(Ws[buf] + nt * 16 * LDH, 0), acc[nt]);
      acc[nt] = wmma_f16(a1, frag_b(Ws[buf] + nt * 16 * LDH, 32), acc[nt]);
    }
    wait_async();
    __syncthreads();
  }
#pragma unroll
  for (int nt = 0; nt < 4; ++nt) {
    int d = nt * 16 + colb;
    float bv_ = bias[h * 64 + d];
#pragma unroll
    for (int e = 0; e < 8; ++e) {
      int m = m0 + wid * 16 + rowb + e;
      int b = m >> 10, s = m & 1023;
      dst[(((size_t)(b * HEADS + h)) * SEQ + s) * HDIM + d] =
          (_Float16)(acc[nt][e] + bv_);
    }
  }
}

// ---------------- k~ = k @ G_h per head (K=64, one tile) ----------------
__global__ void __launch_bounds__(128)
tilde_gemm(const _Float16* __restrict__ Kh, const _Float16* __restrict__ Gh,
           _Float16* __restrict__ Ktl) {
  __shared__ __align__(16) _Float16 Ys[64 * LDH];
  __shared__ __align__(16) _Float16 Gs[64 * LDH];
  const int bh = blockIdx.y;
  const int h  = bh % HEADS;
  const int m0 = blockIdx.x * 64;
  const size_t base = (size_t)bh * SEQ * HDIM;
  async_stage_tile(Ys, Kh + base + (size_t)m0 * HDIM, HDIM);
  async_stage_tile(Gs, Gh + (size_t)h * HDIM * HDIM, HDIM);
  wait_async();
  __syncthreads();
  const int wid  = threadIdx.x >> 5;
  const int lane = threadIdx.x & 31;
  const int colb = lane & 15, rowb = (lane >> 4) * 8;
  v16h a0 = frag_a(Ys + wid * 16 * LDH, 0);
  v16h a1 = frag_a(Ys + wid * 16 * LDH, 32);
  v8f acc[4] = {};
#pragma unroll
  for (int nt = 0; nt < 4; ++nt) {
    acc[nt] = wmma_f16(a0, frag_b(Gs + nt * 16 * LDH, 0), acc[nt]);
    acc[nt] = wmma_f16(a1, frag_b(Gs + nt * 16 * LDH, 32), acc[nt]);
  }
#pragma unroll
  for (int nt = 0; nt < 4; ++nt)
#pragma unroll
    for (int e = 0; e < 8; ++e)
      Ktl[base + (size_t)(m0 + wid * 16 + rowb + e) * HDIM + nt * 16 + colb] =
          (_Float16)acc[nt][e];
}

// ---------------- qq = q^T G q, kk = k^T G k exactly (f32, from A) ----------------
__global__ void qqkk_kernel(const _Float16* __restrict__ Qh,
                            const _Float16* __restrict__ Kh,
                            const float* __restrict__ A,
                            float* __restrict__ qq, float* __restrict__ kk) {
  int idx = blockIdx.x * blockDim.x + threadIdx.x;
  const int total = BH * SEQ;
  if (idx >= 2 * total) return;
  int sel = idx >= total;
  int row = idx - sel * total;
  const _Float16* src = (sel ? Kh : Qh) + (size_t)row * HDIM;
  int h = (row / SEQ) % HEADS;
  const float* Ah = A + (size_t)h * HDIM * RANK;
  float x[HDIM];
  float nrm = 0.f;
#pragma unroll
  for (int d = 0; d < HDIM; ++d) { x[d] = (float)src[d]; nrm += x[d] * x[d]; }
  float acc = EPSV * nrm;
  for (int r = 0; r < RANK; ++r) {
    float s = 0.f;
#pragma unroll
    for (int d = 0; d < HDIM; ++d) s += x[d] * Ah[d * RANK + r];
    acc += s * s;
  }
  (sel ? kk : qq)[row] = acc;
}

// ---------------- flash attention: scores = (2 q·k~ - qq - kk)*SCALE + mask ----------------
__global__ void __launch_bounds__(128)
attn_kernel(const _Float16* __restrict__ Qh, const _Float16* __restrict__ Ktl,
            const _Float16* __restrict__ Vh, const float* __restrict__ qq,
            const float* __restrict__ kk, const float* __restrict__ mask,
            _Float16* __restrict__ Ctx) {
  __shared__ __align__(16) _Float16 Qs[64 * LDH];
  __shared__ __align__(16) _Float16 Ks[64 * LDH];       // k~ tile [t][d]
  __shared__ __align__(16) _Float16 Vs[64 * LDH];       // V^T tile [d][t]
  __shared__ __align__(16) _Float16 Ps[4 * 16 * LDH];   // per-wave P staging
  __shared__ float kks[64];
  __shared__ float msk[64];

  const int bh = blockIdx.y;
  const int b = bh / HEADS, h = bh % HEADS;
  const int s0 = blockIdx.x * 64;
  const size_t base = (size_t)bh * SEQ * HDIM;
  const int wid  = threadIdx.x >> 5;
  const int lane = threadIdx.x & 31;
  const int colb = lane & 15, rowb = (lane >> 4) * 8;

  async_stage_tile(Qs, Qh + base + (size_t)s0 * HDIM, HDIM);  // waited in iter 0

  float qqr[8];
#pragma unroll
  for (int e = 0; e < 8; ++e)
    qqr[e] = qq[bh * SEQ + s0 + wid * 16 + rowb + e];

  float mrun[8], lrun[8];
#pragma unroll
  for (int e = 0; e < 8; ++e) { mrun[e] = -3.0e38f; lrun[e] = 0.f; }
  v8f O[4] = {};

  _Float16* Pw = Ps + wid * 16 * LDH;
  const _Float16* Qw = Qs + wid * 16 * LDH;

  for (int t0 = 0; t0 < SEQ; t0 += 64) {
    __syncthreads();                       // previous tile fully consumed
    async_stage_tile(Ks, Ktl + base + (size_t)t0 * HDIM, HDIM);
    for (int j = 0; j < 32; ++j) {         // V -> LDS transposed [d][t]
      int i = threadIdx.x + 128 * j;
      int t = i >> 6, d = i & 63;
      Vs[d * LDH + t] = Vh[base + (size_t)(t0 + t) * HDIM + d];
    }
    if (threadIdx.x < 64) {
      kks[threadIdx.x] = kk[bh * SEQ + t0 + threadIdx.x];
      msk[threadIdx.x] = mask[b * SEQ + t0 + threadIdx.x];
    }
    if (t0 + 64 < SEQ) {                   // hint next K/V tiles toward caches
      __builtin_prefetch(Ktl + base + (size_t)(t0 + 64 + (threadIdx.x >> 1)) * HDIM, 0, 1);
      __builtin_prefetch(Vh + base + (size_t)(t0 + 64 + (threadIdx.x >> 1)) * HDIM, 0, 1);
    }
    wait_async();
    __syncthreads();

    v16h a0 = frag_a(Qw, 0), a1 = frag_a(Qw, 32);
    float p[4][8], tmax[8];
#pragma unroll
    for (int e = 0; e < 8; ++e) tmax[e] = -3.0e38f;
#pragma unroll
    for (int nt = 0; nt < 4; ++nt) {
      v8f cr = {};
      cr = wmma_f16(a0, frag_b(Ks + nt * 16 * LDH, 0), cr);
      cr = wmma_f16(a1, frag_b(Ks + nt * 16 * LDH, 32), cr);
      float kkc = kks[nt * 16 + colb];
      float mc  = msk[nt * 16 + colb];
#pragma unroll
      for (int e = 0; e < 8; ++e) {
        float sc = (2.f * cr[e] - qqr[e] - kkc) * SCALE + mc;
        p[nt][e] = sc;
        tmax[e] = fmaxf(tmax[e], sc);
      }
    }
    // row max / row sum live in 16-lane halves (C-layout: col = lane&15)
#pragma unroll
    for (int e = 0; e < 8; ++e) {
      float v = tmax[e];
      v = fmaxf(v, __shfl_xor(v, 1, 32));
      v = fmaxf(v, __shfl_xor(v, 2, 32));
      v = fmaxf(v, __shfl_xor(v, 4, 32));
      v = fmaxf(v, __shfl_xor(v, 8, 32));
      float nm = fmaxf(mrun[e], v);
      float scal = __expf(mrun[e] - nm);
      mrun[e] = nm;
      lrun[e] *= scal;
#pragma unroll
      for (int nt = 0; nt < 4; ++nt) O[nt][e] *= scal;
    }
    float rsum[8];
#pragma unroll
    for (int e = 0; e < 8; ++e) rsum[e] = 0.f;
#pragma unroll
    for (int nt = 0; nt < 4; ++nt)
#pragma unroll
      for (int e = 0; e < 8; ++e) {
        float pe = __expf(p[nt][e] - mrun[e]);
        rsum[e] += pe;
        Pw[(rowb + e) * LDH + nt * 16 + colb] = (_Float16)pe;  // C-layout -> A-layout via LDS
      }
#pragma unroll
    for (int e = 0; e < 8; ++e) {
      float v = rsum[e];
      v += __shfl_xor(v, 1, 32);
      v += __shfl_xor(v, 2, 32);
      v += __shfl_xor(v, 4, 32);
      v += __shfl_xor(v, 8, 32);
      lrun[e] += v;
    }
    __syncthreads();                       // make P stores visible / no reorder
    v16h pa0 = frag_a(Pw, 0), pa1 = frag_a(Pw, 32);
#pragma unroll
    for (int nt = 0; nt < 4; ++nt) {
      O[nt] = wmma_f16(pa0, frag_b(Vs + nt * 16 * LDH, 0), O[nt]);
      O[nt] = wmma_f16(pa1, frag_b(Vs + nt * 16 * LDH, 32), O[nt]);
    }
  }

#pragma unroll
  for (int nt = 0; nt < 4; ++nt)
#pragma unroll
    for (int e = 0; e < 8; ++e) {
      int srow = s0 + wid * 16 + rowb + e;
      Ctx[((size_t)(b * SEQ + srow)) * EMBED + h * HDIM + nt * 16 + colb] =
          (_Float16)(O[nt][e] / lrun[e]);
    }
}

// ---------------- output projection: out = ctx @ Wo^T + bo (f32 out) ----------------
__global__ void __launch_bounds__(128)
gemm_out(const _Float16* __restrict__ Ctx, const _Float16* __restrict__ Who,
         const float* __restrict__ bo, float* __restrict__ out) {
  __shared__ __align__(16) _Float16 Xs[2][64 * LDH];
  __shared__ __align__(16) _Float16 Ws[2][64 * LDH];
  const int n0 = blockIdx.x * 64;
  const int m0 = blockIdx.y * 64;
  const int wid  = threadIdx.x >> 5;
  const int lane = threadIdx.x & 31;
  const int colb = lane & 15, rowb = (lane >> 4) * 8;

  async_stage_tile(Xs[0], Ctx + (size_t)m0 * EMBED, EMBED);
  async_stage_tile(Ws[0], Who + (size_t)n0 * EMBED, EMBED);
  wait_async();
  __syncthreads();

  v8f acc[4] = {};
#pragma unroll
  for (int kb = 0; kb < EMBED; kb += 64) {
    const int buf = (kb >> 6) & 1;
    if (kb + 64 < EMBED) {
      async_stage_tile(Xs[buf ^ 1], Ctx + (size_t)m0 * EMBED + kb + 64, EMBED);
      async_stage_tile(Ws[buf ^ 1], Who + (size_t)n0 * EMBED + kb + 64, EMBED);
    }
    v16h a0 = frag_a(Xs[buf] + wid * 16 * LDH, 0);
    v16h a1 = frag_a(Xs[buf] + wid * 16 * LDH, 32);
#pragma unroll
    for (int nt = 0; nt < 4; ++nt) {
      acc[nt] = wmma_f16(a0, frag_b(Ws[buf] + nt * 16 * LDH, 0), acc[nt]);
      acc[nt] = wmma_f16(a1, frag_b(Ws[buf] + nt * 16 * LDH, 32), acc[nt]);
    }
    wait_async();
    __syncthreads();
  }
#pragma unroll
  for (int nt = 0; nt < 4; ++nt) {
    int n = n0 + nt * 16 + colb;
    float bv_ = bo[n];
#pragma unroll
    for (int e = 0; e < 8; ++e) {
      int m = m0 + wid * 16 + rowb + e;
      out[(size_t)m * EMBED + n] = acc[nt][e] + bv_;
    }
  }
}

extern "C" void kernel_launch(void* const* d_in, const int* in_sizes, int n_in,
                              void* d_out, int out_size, void* d_ws, size_t ws_size,
                              hipStream_t stream) {
  const float* hidden = (const float*)d_in[0];
  const float* mask   = (const float*)d_in[1];
  const float* Wq = (const float*)d_in[2];
  const float* bq = (const float*)d_in[3];
  const float* Wk = (const float*)d_in[4];
  const float* bk = (const float*)d_in[5];
  const float* Wv = (const float*)d_in[6];
  const float* bv = (const float*)d_in[7];
  const float* Wo = (const float*)d_in[8];
  const float* bo = (const float*)d_in[9];
  const float* A  = (const float*)d_in[10];
  float* out = (float*)d_out;

  char* ws = (char*)d_ws;
  size_t off = 0;
  auto take = [&](size_t bytes) -> char* {
    char* p = ws + off;
    off += (bytes + 255) & ~(size_t)255;
    return p;
  };
  _Float16* Xh  = (_Float16*)take((size_t)MROWS * EMBED * 2);
  _Float16* Whq = (_Float16*)take((size_t)EMBED * EMBED * 2);
  _Float16* Whk = (_Float16*)take((size_t)EMBED * EMBED * 2);
  _Float16* Whv = (_Float16*)take((size_t)EMBED * EMBED * 2);
  _Float16* Who = (_Float16*)take((size_t)EMBED * EMBED * 2);
  _Float16* Gh  = (_Float16*)take((size_t)HEADS * HDIM * HDIM * 2);
  _Float16* Qh  = (_Float16*)take((size_t)BH * SEQ * HDIM * 2);
  _Float16* Kh  = (_Float16*)take((size_t)BH * SEQ * HDIM * 2);
  _Float16* Vh  = (_Float16*)take((size_t)BH * SEQ * HDIM * 2);
  _Float16* Ktl = (_Float16*)take((size_t)BH * SEQ * HDIM * 2);
  float*    qqb = (float*)take((size_t)BH * SEQ * 4);
  float*    kkb = (float*)take((size_t)BH * SEQ * 4);
  _Float16* Ctx = (_Float16*)take((size_t)MROWS * EMBED * 2);
  // total workspace ~43 MB

  cvt_f32_f16<<<(MROWS * EMBED + 255) / 256, 256, 0, stream>>>(hidden, Xh, MROWS * EMBED);
  cvt_f32_f16<<<(EMBED * EMBED + 255) / 256, 256, 0, stream>>>(Wq, Whq, EMBED * EMBED);
  cvt_f32_f16<<<(EMBED * EMBED + 255) / 256, 256, 0, stream>>>(Wk, Whk, EMBED * EMBED);
  cvt_f32_f16<<<(EMBED * EMBED + 255) / 256, 256, 0, stream>>>(Wv, Whv, EMBED * EMBED);
  cvt_f32_f16<<<(EMBED * EMBED + 255) / 256, 256, 0, stream>>>(Wo, Who, EMBED * EMBED);
  build_g<<<HEADS, 256, 0, stream>>>(A, Gh);
  gemm_qkv<<<dim3(HEADS, MROWS / 64, 3), 128, 0, stream>>>(Xh, Whq, Whk, Whv,
                                                           bq, bk, bv, Qh, Kh, Vh);
  tilde_gemm<<<dim3(SEQ / 64, BH), 128, 0, stream>>>(Kh, Gh, Ktl);
  qqkk_kernel<<<(2 * BH * SEQ + 255) / 256, 256, 0, stream>>>(Qh, Kh, A, qqb, kkb);
  attn_kernel<<<dim3(SEQ / 64, BH), 128, 0, stream>>>(Qh, Ktl, Vh, qqb, kkb, mask, Ctx);
  gemm_out<<<dim3(EMBED / 64, MROWS / 64), 128, 0, stream>>>(Ctx, Who, bo, out);
}
